// MultiHeadConvAttention_79920751443981
// MI455X (gfx1250) — compile-verified, matrix operand order checked by reading, not run on hardware
//
#include <hip/hip_runtime.h>
#include <hip/hip_bf16.h>

typedef __bf16 bf16;
typedef __attribute__((ext_vector_type(16))) __bf16 v16bf;
typedef __attribute__((ext_vector_type(8)))  float  v8f;
typedef int v4i __attribute__((vector_size(16)));   // matches builtin prototype

#define AS1 __attribute__((address_space(1)))
#define AS3 __attribute__((address_space(3)))

#if defined(__AMDGCN__) && __has_builtin(__builtin_amdgcn_global_load_async_to_lds_b128)
#define HAVE_ASYNC_LDS 1
#else
#define HAVE_ASYNC_LDS 0
#endif

static constexpr int BB = 16;        // batch
static constexpr int SS = 1024;      // sequence
static constexpr int DMODEL = 512;   // d_model == feature_len
static constexpr int NH = 8;         // heads
static constexpr int DHEAD = 64;     // d_k == d_v
static constexpr size_t LN_ELEMS = (size_t)BB * SS * DMODEL;        // 8,388,608
static constexpr size_t ATTN_ELEMS = (size_t)BB * NH * SS * SS;     // 134,217,728

static __device__ __forceinline__ bf16 f2bf(float x) { return (bf16)x; }

// Async memory->LDS copy of 16 bytes (ASYNCcnt path); falls back to a
// VGPR round-trip copy when the builtin is unavailable.
static __device__ __forceinline__ void async_copy_b128(const void* g, void* l) {
#if HAVE_ASYNC_LDS
  __builtin_amdgcn_global_load_async_to_lds_b128(
      (AS1 v4i*)(uintptr_t)g,
      (AS3 v4i*)(uint32_t)(uintptr_t)l, 0, 0);
#else
  *reinterpret_cast<uint4*>(l) = *reinterpret_cast<const uint4*>(g);
#endif
}

static __device__ __forceinline__ void async_wait_all() {
#if HAVE_ASYNC_LDS
#if __has_builtin(__builtin_amdgcn_s_wait_asynccnt)
  __builtin_amdgcn_s_wait_asynccnt(0);
#else
  asm volatile("s_wait_asynccnt 0x0" ::: "memory");
#endif
#endif
}

union FragU { uint4 u[2]; v16bf v; };

// Load a 16-lane bf16 WMMA operand fragment from an operand-major LDS tile.
// Row r holds 16 rows (M or N) selected by (lane&15); per-lane K elements are
// {e0..e0+7, e0+16..e0+23}. Row byte stride (LDW*2) must be a multiple of 16.
template <int LDW>
static __device__ __forceinline__ v16bf load_frag16(const bf16* tile, int row, int e0) {
  const bf16* rp = tile + row * LDW + e0;
  const uint4* p = reinterpret_cast<const uint4*>(rp);
  FragU f;
  f.u[0] = p[0];
  f.u[1] = p[2];   // +32 bytes == +16 bf16 elements
  return f.v;
}

// ---------------------------------------------------------------------------
// 3x3 single-channel "conv2d" over [B, S, F] with zero padding, bf16 output.
// ---------------------------------------------------------------------------
__global__ __launch_bounds__(256) void conv3x3_bf16(
    const float* __restrict__ x, const float* __restrict__ w9,
    const float* __restrict__ bias, bf16* __restrict__ y) {
  size_t idx = (size_t)blockIdx.x * 256 + threadIdx.x;   // < B*S*F
  int f = (int)(idx % DMODEL);
  int s = (int)((idx / DMODEL) % SS);
  int b = (int)(idx / ((size_t)DMODEL * SS));
  const float* xb = x + (size_t)b * SS * DMODEL;
  float acc = bias[0];
#pragma unroll
  for (int di = -1; di <= 1; ++di) {
    int ss = s + di;
    if (ss < 0 || ss >= SS) continue;
#pragma unroll
    for (int dj = -1; dj <= 1; ++dj) {
      int ff = f + dj;
      if (ff < 0 || ff >= DMODEL) continue;
      acc += w9[(di + 1) * 3 + (dj + 1)] * xb[(size_t)ss * DMODEL + ff];
    }
  }
  y[idx] = f2bf(acc);
}

__global__ __launch_bounds__(256) void cvt_f32_bf16(
    const float* __restrict__ src, bf16* __restrict__ dst, int n) {
  int i = blockIdx.x * 256 + threadIdx.x;
  if (i < n) dst[i] = f2bf(src[i]);
}

// ---------------------------------------------------------------------------
// Generic row-major bf16 GEMM: C[M,N] = A[M,K] * B[K,N]; WMMA bf16, f32 accum.
// Block: 256 threads (8 waves), tile 128x128, BK=32. Wave (wm,wn) in 4x2 grid
// owns a 32x64 region = 2x4 16x16 WMMA tiles. A tile is staged via async
// memory->LDS DMA; B tile is transposed to N-major during staging.
// ---------------------------------------------------------------------------
template <bool OUT_BF16>
__global__ __launch_bounds__(256) void gemm_bf16(
    const bf16* __restrict__ A, const bf16* __restrict__ Bm,
    void* __restrict__ Cout, int M, int N, int K, int lda, int ldb, int ldc) {
  constexpr int BK = 32, LDW = BK + 8;   // row stride 40 bf16 = 80 B (16B mult.)
  __shared__ bf16 As[128 * LDW];         // M-major
  __shared__ bf16 Bs[128 * LDW];         // N-major (transposed on store)
  const int tid = threadIdx.x;
  const int lane = tid & 31, w = tid >> 5;
  const int lr = lane & 15, hi = (lane >> 4) * 8;
  const int wm = w >> 1, wn = w & 1;
  const int gm0 = blockIdx.y * 128, gn0 = blockIdx.x * 128;

  const int ar = tid >> 1, ac = (tid & 1) * 16;     // A staging coords
  const int br = tid >> 3, bc = (tid & 7) * 16;     // B staging coords

  v8f acc[2][4] = {};

  for (int k0 = 0; k0 < K; k0 += BK) {
    {  // A tile 128x32: 16 bf16 per thread, async to LDS
      const bf16* g = A + (size_t)(gm0 + ar) * lda + k0 + ac;
      bf16* s = &As[ar * LDW + ac];
      async_copy_b128(g, s);
      async_copy_b128(g + 8, s + 8);
    }
    {  // B tile 32x128 (K-major in memory) -> Bs[n][k]
      const uint4* g = reinterpret_cast<const uint4*>(Bm + (size_t)(k0 + br) * ldb + gn0 + bc);
      uint4 d0 = g[0], d1 = g[1];
      const bf16* e0 = reinterpret_cast<const bf16*>(&d0);
      const bf16* e1 = reinterpret_cast<const bf16*>(&d1);
#pragma unroll
      for (int i = 0; i < 8; ++i) Bs[(bc + i) * LDW + br] = e0[i];
#pragma unroll
      for (int i = 0; i < 8; ++i) Bs[(bc + 8 + i) * LDW + br] = e1[i];
    }
    if (k0 + BK < K) {  // cover L2->WGP latency for the next tiles
      __builtin_prefetch(A + (size_t)(gm0 + ar) * lda + k0 + BK + ac, 0, 3);
      __builtin_prefetch(Bm + (size_t)(k0 + BK + br) * ldb + gn0 + bc, 0, 3);
    }
    async_wait_all();
    __syncthreads();

    v16bf af[2], bq[4];
#pragma unroll
    for (int mi = 0; mi < 2; ++mi)
      af[mi] = load_frag16<LDW>(As, wm * 32 + mi * 16 + lr, hi);
#pragma unroll
    for (int ni = 0; ni < 4; ++ni)
      bq[ni] = load_frag16<LDW>(Bs, wn * 64 + ni * 16 + lr, hi);
#pragma unroll
    for (int mi = 0; mi < 2; ++mi)
#pragma unroll
      for (int ni = 0; ni < 4; ++ni)
        acc[mi][ni] = __builtin_amdgcn_wmma_f32_16x16x32_bf16(
            false, af[mi], false, bq[ni], (short)0, acc[mi][ni], false, false);
    __syncthreads();
  }

  // D layout: VGPR i -> M = (lane>=16 ? 8 : 0) + i, N = lane&15
#pragma unroll
  for (int mi = 0; mi < 2; ++mi)
#pragma unroll
    for (int ni = 0; ni < 4; ++ni) {
      int m = gm0 + wm * 32 + mi * 16 + hi;
      int n = gn0 + wn * 64 + ni * 16 + lr;
#pragma unroll
      for (int i = 0; i < 8; ++i) {
        float v = acc[mi][ni][i];
        if (OUT_BF16) ((bf16*)Cout)[(size_t)(m + i) * ldc + n] = f2bf(v);
        else          ((float*)Cout)[(size_t)(m + i) * ldc + n] = v;
      }
    }
}

// ---------------------------------------------------------------------------
// scores[b,h,q,k] = 0.125 * dot(Q[b,q,h,:], K[b,k,h,:]); f32 out to d_out.
// Per-(b,h) GEMM, M=N=1024, K=64. Tile 128x128, full-K staged once in LDS via
// async memory->LDS DMA (both operands are straight copies here).
// ---------------------------------------------------------------------------
__global__ __launch_bounds__(256) void attn_scores(
    const bf16* __restrict__ Q, const bf16* __restrict__ Kmat,
    float* __restrict__ scores) {
  constexpr int LDW = 72;               // 64 + 8 -> 144 B row stride
  __shared__ bf16 Qs[128 * LDW];
  __shared__ bf16 Ks[128 * LDW];
  const int tid = threadIdx.x;
  const int lane = tid & 31, w = tid >> 5;
  const int lr = lane & 15, hi = (lane >> 4) * 8;
  const int wm = w >> 1, wn = w & 1;
  const int bh = blockIdx.z, b = bh >> 3, h = bh & 7;
  const int q0 = blockIdx.y * 128, n0 = blockIdx.x * 128;
  const bf16* qbase = Q + (size_t)b * SS * DMODEL + h * DHEAD;
  const bf16* kbase = Kmat + (size_t)b * SS * DMODEL + h * DHEAD;
  {
    int r = tid >> 1, c = (tid & 1) * 32;
    const bf16* gq = qbase + (size_t)(q0 + r) * DMODEL + c;
    const bf16* gk = kbase + (size_t)(n0 + r) * DMODEL + c;
    bf16* sq = &Qs[r * LDW + c];
    bf16* sk = &Ks[r * LDW + c];
#pragma unroll
    for (int i = 0; i < 4; ++i) {
      async_copy_b128(gq + i * 8, sq + i * 8);
      async_copy_b128(gk + i * 8, sk + i * 8);
    }
  }
  async_wait_all();
  __syncthreads();

  v8f acc[2][4] = {};
#pragma unroll
  for (int kk = 0; kk < DHEAD; kk += 32) {
    v16bf af[2], bq[4];
#pragma unroll
    for (int mi = 0; mi < 2; ++mi)
      af[mi] = load_frag16<LDW>(Qs, wm * 32 + mi * 16 + lr, kk + hi);
#pragma unroll
    for (int ni = 0; ni < 4; ++ni)
      bq[ni] = load_frag16<LDW>(Ks, wn * 64 + ni * 16 + lr, kk + hi);
#pragma unroll
    for (int mi = 0; mi < 2; ++mi)
#pragma unroll
      for (int ni = 0; ni < 4; ++ni)
        acc[mi][ni] = __builtin_amdgcn_wmma_f32_16x16x32_bf16(
            false, af[mi], false, bq[ni], (short)0, acc[mi][ni], false, false);
  }

  size_t base = (size_t)bh * SS * SS;
#pragma unroll
  for (int mi = 0; mi < 2; ++mi)
#pragma unroll
    for (int ni = 0; ni < 4; ++ni) {
      int m = q0 + wm * 32 + mi * 16 + hi;
      int n = n0 + wn * 64 + ni * 16 + lr;
#pragma unroll
      for (int i = 0; i < 8; ++i)
        scores[base + (size_t)(m + i) * SS + n] = acc[mi][ni][i] * 0.125f;
    }
}

// ---------------------------------------------------------------------------
// In-place masked row softmax over the last axis (length 1024).
// grid = (S, H, B), 256 threads, 4 elements per thread.
// ---------------------------------------------------------------------------
__global__ __launch_bounds__(256) void softmax_rows(
    float* __restrict__ attn, const unsigned char* __restrict__ mask) {
  const int q = blockIdx.x, h = blockIdx.y, b = blockIdx.z;
  const int tid = threadIdx.x, lane = tid & 31, w = tid >> 5;
  size_t base = (((size_t)(b * NH + h)) * SS + q) * SS;
  const unsigned char* mrow = mask + ((size_t)b * SS + q) * SS;
  __shared__ float red[8];

  float v[4];
  float mx = -3.4e38f;
#pragma unroll
  for (int i = 0; i < 4; ++i) {
    int idx = tid + i * 256;
    float s = attn[base + idx];
    if (mrow[idx]) s = -1e9f;
    v[i] = s;
    mx = fmaxf(mx, s);
  }
#pragma unroll
  for (int o = 16; o > 0; o >>= 1) mx = fmaxf(mx, __shfl_xor(mx, o, 32));
  if (lane == 0) red[w] = mx;
  __syncthreads();
  mx = red[0];
#pragma unroll
  for (int i = 1; i < 8; ++i) mx = fmaxf(mx, red[i]);
  __syncthreads();

  float sum = 0.0f;
#pragma unroll
  for (int i = 0; i < 4; ++i) {
    v[i] = __expf(v[i] - mx);
    sum += v[i];
  }
#pragma unroll
  for (int o = 16; o > 0; o >>= 1) sum += __shfl_xor(sum, o, 32);
  if (lane == 0) red[w] = sum;
  __syncthreads();
  sum = red[0] + red[1] + red[2] + red[3] + red[4] + red[5] + red[6] + red[7];
  float inv = 1.0f / sum;
#pragma unroll
  for (int i = 0; i < 4; ++i) attn[base + tid + i * 256] = v[i] * inv;
}

// ---------------------------------------------------------------------------
// context[b,q,h,:] = attn[b,h,q,:] @ V[b,:,h,:]. Per-(b,h) GEMM,
// M=1024, N=64, K=1024. attn is f32 in d_out; converted to bf16 into LDS.
// ---------------------------------------------------------------------------
__global__ __launch_bounds__(256) void attn_context(
    const float* __restrict__ attn, const bf16* __restrict__ V,
    bf16* __restrict__ ctx) {
  constexpr int LDW = 40;
  __shared__ bf16 As[128 * LDW];   // attn tile, q-major
  __shared__ bf16 Vs[64 * LDW];    // V tile, d-major (transposed on store)
  const int tid = threadIdx.x;
  const int lane = tid & 31, w = tid >> 5;
  const int lr = lane & 15, hi = (lane >> 4) * 8;
  const int bh = blockIdx.z, b = bh >> 3, h = bh & 7;
  const int q0 = blockIdx.y * 128;
  const size_t abase = (size_t)bh * SS * SS;
  const bf16* vbase = V + (size_t)b * SS * DMODEL + h * DHEAD;

  v8f acc[4] = {};
  for (int k0 = 0; k0 < SS; k0 += 32) {
    {  // attn tile 128x32 f32 -> bf16
      int r = tid >> 1, c = (tid & 1) * 16;
      const float4* g = reinterpret_cast<const float4*>(attn + abase + (size_t)(q0 + r) * SS + k0 + c);
      float fv[16];
      *reinterpret_cast<float4*>(&fv[0])  = g[0];
      *reinterpret_cast<float4*>(&fv[4])  = g[1];
      *reinterpret_cast<float4*>(&fv[8])  = g[2];
      *reinterpret_cast<float4*>(&fv[12]) = g[3];
      bf16* s = &As[r * LDW + c];
#pragma unroll
      for (int i = 0; i < 16; ++i) s[i] = f2bf(fv[i]);
    }
    {  // V tile 32x64 -> Vs[d][k]
      int r = tid >> 3, c = (tid & 7) * 8;
      uint4 d0 = *reinterpret_cast<const uint4*>(vbase + (size_t)(k0 + r) * DMODEL + c);
      const bf16* e = reinterpret_cast<const bf16*>(&d0);
#pragma unroll
      for (int i = 0; i < 8; ++i) Vs[(c + i) * LDW + r] = e[i];
    }
    __syncthreads();

    v16bf af = load_frag16<LDW>(As, w * 16 + lr, hi);
#pragma unroll
    for (int ni = 0; ni < 4; ++ni) {
      v16bf bq = load_frag16<LDW>(Vs, ni * 16 + lr, hi);
      acc[ni] = __builtin_amdgcn_wmma_f32_16x16x32_bf16(
          false, af, false, bq, (short)0, acc[ni], false, false);
    }
    __syncthreads();
  }

  bf16* cbase = ctx + (size_t)b * SS * DMODEL + h * DHEAD;
#pragma unroll
  for (int ni = 0; ni < 4; ++ni)
#pragma unroll
    for (int i = 0; i < 8; ++i)
      cbase[(size_t)(q0 + w * 16 + hi + i) * DMODEL + ni * 16 + lr] = f2bf(acc[ni][i]);
}

// ---------------------------------------------------------------------------
// out = LayerNorm(fcout + residual) over last dim (512). grid = B*S rows.
// ---------------------------------------------------------------------------
__global__ __launch_bounds__(256) void layernorm_res(
    const float* __restrict__ fcout, const float* __restrict__ resid,
    float* __restrict__ out) {
  const int row = blockIdx.x;
  const int tid = threadIdx.x, lane = tid & 31, w = tid >> 5;
  const size_t base = (size_t)row * DMODEL;
  __shared__ float red[8];

  float a = fcout[base + tid] + resid[base + tid];
  float b2 = fcout[base + tid + 256] + resid[base + tid + 256];

  float sum = a + b2;
#pragma unroll
  for (int o = 16; o > 0; o >>= 1) sum += __shfl_xor(sum, o, 32);
  if (lane == 0) red[w] = sum;
  __syncthreads();
  sum = red[0] + red[1] + red[2] + red[3] + red[4] + red[5] + red[6] + red[7];
  float mu = sum * (1.0f / DMODEL);
  __syncthreads();

  float da = a - mu, db = b2 - mu;
  float vs = da * da + db * db;
#pragma unroll
  for (int o = 16; o > 0; o >>= 1) vs += __shfl_xor(vs, o, 32);
  if (lane == 0) red[w] = vs;
  __syncthreads();
  vs = red[0] + red[1] + red[2] + red[3] + red[4] + red[5] + red[6] + red[7];
  float r = rsqrtf(vs * (1.0f / DMODEL) + 1e-5f);

  out[base + tid] = da * r;
  out[base + tid + 256] = db * r;
}

// ---------------------------------------------------------------------------
extern "C" void kernel_launch(void* const* d_in, const int* in_sizes, int n_in,
                              void* d_out, int out_size, void* d_ws, size_t ws_size,
                              hipStream_t stream) {
  const float* inQ = (const float*)d_in[0];
  const float* inK = (const float*)d_in[1];
  const float* inV = (const float*)d_in[2];
  const unsigned char* mask = (const unsigned char*)d_in[3];
  const float* cQw = (const float*)d_in[4];
  const float* cQb = (const float*)d_in[5];
  const float* cKw = (const float*)d_in[6];
  const float* cKb = (const float*)d_in[7];
  const float* cVw = (const float*)d_in[8];
  const float* cVb = (const float*)d_in[9];
  const float* WQ = (const float*)d_in[10];
  const float* WK = (const float*)d_in[11];
  const float* WV = (const float*)d_in[12];
  const float* FC = (const float*)d_in[13];
  float* out = (float*)d_out;
  float* attn_out = out + LN_ELEMS;

  char* ws = (char*)d_ws;
  size_t off = 0;
  const size_t tens_b = LN_ELEMS * sizeof(bf16);       // 16 MiB per [B,S,512] bf16
  const size_t wmat_b = (size_t)DMODEL * DMODEL * sizeof(bf16);
  bf16* qc = (bf16*)(ws + off); off += tens_b;
  bf16* kc = (bf16*)(ws + off); off += tens_b;
  bf16* vc = (bf16*)(ws + off); off += tens_b;
  bf16* wq = (bf16*)(ws + off); off += wmat_b;
  bf16* wk = (bf16*)(ws + off); off += wmat_b;
  bf16* wv = (bf16*)(ws + off); off += wmat_b;
  bf16* wfc = (bf16*)(ws + off); off += wmat_b;
  bf16* Qp = (bf16*)(ws + off); off += tens_b;
  bf16* Kp = (bf16*)(ws + off); off += tens_b;
  bf16* Vp = (bf16*)(ws + off); off += tens_b;
  bf16* ctx = (bf16*)(ws + off); off += tens_b;
  float* fco = (float*)(ws + off); off += LN_ELEMS * sizeof(float);

  const dim3 blk(256);
  const int nconv = (int)(LN_ELEMS / 256);             // 32768 blocks
  conv3x3_bf16<<<nconv, blk, 0, stream>>>(inQ, cQw, cQb, qc);
  conv3x3_bf16<<<nconv, blk, 0, stream>>>(inK, cKw, cKb, kc);
  conv3x3_bf16<<<nconv, blk, 0, stream>>>(inV, cVw, cVb, vc);

  const int nw = DMODEL * DMODEL;                      // 262144
  cvt_f32_bf16<<<nw / 256, blk, 0, stream>>>(WQ, wq, nw);
  cvt_f32_bf16<<<nw / 256, blk, 0, stream>>>(WK, wk, nw);
  cvt_f32_bf16<<<nw / 256, blk, 0, stream>>>(WV, wv, nw);
  cvt_f32_bf16<<<nw / 256, blk, 0, stream>>>(FC, wfc, nw);

  const int M = BB * SS;                               // 16384
  dim3 ggrid(DMODEL / 128, M / 128);                   // (4, 128)
  gemm_bf16<true><<<ggrid, blk, 0, stream>>>(qc, wq, Qp, M, DMODEL, DMODEL, DMODEL, DMODEL, DMODEL);
  gemm_bf16<true><<<ggrid, blk, 0, stream>>>(kc, wk, Kp, M, DMODEL, DMODEL, DMODEL, DMODEL, DMODEL);
  gemm_bf16<true><<<ggrid, blk, 0, stream>>>(vc, wv, Vp, M, DMODEL, DMODEL, DMODEL, DMODEL, DMODEL);

  attn_scores<<<dim3(SS / 128, SS / 128, BB * NH), blk, 0, stream>>>(Qp, Kp, attn_out);
  softmax_rows<<<dim3(SS, NH, BB), blk, 0, stream>>>(attn_out, mask);
  attn_context<<<dim3(1, SS / 128, BB * NH), blk, 0, stream>>>(attn_out, Vp, ctx);

  gemm_bf16<false><<<ggrid, blk, 0, stream>>>(ctx, wfc, fco, M, DMODEL, DMODEL, DMODEL, DMODEL, DMODEL);
  layernorm_res<<<M, blk, 0, stream>>>(fco, inQ, out);
}
